// JKGATConvNet_42262478192814
// MI455X (gfx1250) — matrix operand order repaired and checked by reading, not compiled, not run on hardware
//
#include <hip/hip_runtime.h>
#include <hip/hip_bf16.h>
#include <stdint.h>

// ---------------------------------------------------------------------------
// JK-GAT network for MI455X (gfx1250, wave32, WMMA bf16, TDM LDS staging)
// ---------------------------------------------------------------------------

typedef __attribute__((ext_vector_type(16))) __bf16          v16bf;
typedef __attribute__((ext_vector_type(16))) unsigned short  v16us;
typedef __attribute__((ext_vector_type(8)))  float           v8f;
typedef __attribute__((ext_vector_type(4)))  unsigned int    v4u;
typedef __attribute__((ext_vector_type(8)))  int             v8i;
typedef __attribute__((ext_vector_type(4)))  int             v4i;

__device__ __forceinline__ unsigned short f2bfu(float f) {
  unsigned u = __float_as_uint(f);
  u += 0x7FFFu + ((u >> 16) & 1u);          // round-to-nearest-even
  return (unsigned short)(u >> 16);
}
__device__ __forceinline__ __bf16 f2bf(float f) {
  unsigned short s = f2bfu(f);
  return __builtin_bit_cast(__bf16, s);
}
__device__ __forceinline__ v8f zero8() {
  v8f z = {0.f, 0.f, 0.f, 0.f, 0.f, 0.f, 0.f, 0.f};
  return z;
}

// A fragment 16x32 bf16 from an f32 row pointer.
// lane: row = lane&15, half = lane>>4. element e -> k = k0+half*16+e
// (contiguous, matching loadB_bf)
__device__ __forceinline__ v16bf loadA_f32(const float* rowp, int k0, int half) {
  const float* p = rowp + k0 + half * 16;
  v16us u;
#pragma unroll
  for (int i = 0; i < 8; ++i) u[i] = f2bfu(p[i]);
#pragma unroll
  for (int i = 0; i < 8; ++i) u[8 + i] = f2bfu(p[8 + i]);
  return __builtin_bit_cast(v16bf, u);
}

// B fragment 32x16 bf16: lane col = lane&15, khalf = lane>>4; element e -> K=khalf*16+e.
// Weights are stored "B-transposed": Wt[col][k] contiguous in k -> one 32B read.
struct U32x8 { uint4 a, b; };
__device__ __forceinline__ v16bf loadB_bf(const __bf16* p) {
  U32x8 u;
  u.a = *(const uint4*)p;
  u.b = *(const uint4*)(p + 8);
  return __builtin_bit_cast(v16bf, u);
}

__device__ __forceinline__ v8f wmma_bf(v16bf a, v16bf b, v8f c) {
  return __builtin_amdgcn_wmma_f32_16x16x32_bf16(false, a, false, b, (short)0, c,
                                                 false, false);
}

__device__ __forceinline__ float sigm(float x) { return 1.f / (1.f + expf(-x)); }
__device__ __forceinline__ float lrelu(float x) { return x > 0.f ? x : 0.2f * x; }
// order-preserving float<->int map for integer atomicMax
__device__ __forceinline__ int   fmap(float x) { int b = __float_as_int(x); return b >= 0 ? b : (b ^ 0x7FFFFFFF); }
__device__ __forceinline__ float funmap(int b) { int r = b >= 0 ? b : (b ^ 0x7FFFFFFF); return __int_as_float(r); }

// ---------------------------------------------------------------------------
// prep: bf16 weight conversion (+ transpose for W1/W2, zero-pad for W_out)
// ---------------------------------------------------------------------------
__global__ void k_prep(const float* W1, const float* W2,
                       const float* WihF, const float* WhhF,
                       const float* WihB, const float* WhhB,
                       const float* Wout,
                       __bf16* w1t, __bf16* w2t,
                       __bf16* wihF, __bf16* whhF, __bf16* wihB, __bf16* whhB,
                       __bf16* woutp) {
  int t = blockIdx.x * blockDim.x + threadIdx.x;
  if (t < 64 * 512) { int c = t >> 9, k = t & 511; w1t[t] = f2bf(W1[k * 64 + c]); return; }
  t -= 64 * 512;
  if (t < 64 * 64) { int c = t >> 6, k = t & 63; w2t[t] = f2bf(W2[k * 64 + c]); return; }
  t -= 64 * 64;
  if (t < 16384) { wihF[t] = f2bf(WihF[t]); return; } t -= 16384;
  if (t < 16384) { whhF[t] = f2bf(WhhF[t]); return; } t -= 16384;
  if (t < 16384) { wihB[t] = f2bf(WihB[t]); return; } t -= 16384;
  if (t < 16384) { whhB[t] = f2bf(WhhB[t]); return; } t -= 16384;
  if (t < 48 * 64) { int c = t >> 6; woutp[t] = (c < 40) ? f2bf(Wout[t]) : f2bf(0.f); }
}

// ---------------------------------------------------------------------------
// GEMM: H[n,64] = X[n,K] @ W (W given bf16 transposed as Wt[64][K]).
// Wt staged to LDS via the Tensor Data Mover (1-D bf16 tile).
// 256 threads = 8 waves; each wave: 16 rows x 64 cols, K loop step 32.
// ---------------------------------------------------------------------------
__global__ void k_gemm(const float* X, const __bf16* Wt, float* Hm, int nrows, int K) {
  extern __shared__ char smem[];
  const unsigned nelem = 64u * (unsigned)K;   // bf16 elements, <= 32768
  if (threadIdx.x < 32) {
#if __has_builtin(__builtin_amdgcn_tensor_load_to_lds)
    unsigned ldsoff = (unsigned)(size_t)(void*)smem;   // low 32 bits = LDS byte addr
    unsigned long long ga = (unsigned long long)(size_t)(const void*)Wt;
    v4u g0;
    g0[0] = 1u;                                         // count = 1 valid descriptor
    g0[1] = ldsoff;                                     // lds_addr
    g0[2] = (unsigned)(ga & 0xFFFFFFFFull);             // global_addr[31:0]
    g0[3] = (unsigned)((ga >> 32) & 0x01FFFFFFull) | (2u << 30);  // addr[56:32], type=2
    v8i g1;
    g1[0] = (int)(1u << 16);                            // data_size = 2 bytes
    g1[1] = (int)((nelem & 0xFFFFu) << 16);             // tensor_dim0[15:0]
    g1[2] = (int)((nelem >> 16) | (1u << 16));          // tensor_dim0 hi, tensor_dim1=1
    g1[3] = (int)((nelem & 0xFFFFu) << 16);             // tile_dim0
    g1[4] = 0;                                          // tile_dim1/2 unused (1-D)
    g1[5] = (int)nelem;                                 // tensor_dim0_stride lo
    g1[6] = 0;
    g1[7] = 0;
    v4i gz4 = {0, 0, 0, 0};
    v8i gz8 = {0, 0, 0, 0, 0, 0, 0, 0};                 // unused extra group (6-arg form)
    __builtin_amdgcn_tensor_load_to_lds(g0, g1, gz4, gz4, gz8, 0);
    __builtin_amdgcn_s_wait_tensorcnt(0);
#else
    for (unsigned i = threadIdx.x; i < nelem / 8u; i += 32u)
      ((uint4*)smem)[i] = ((const uint4*)Wt)[i];
#endif
  }
  __syncthreads();
  const __bf16* Wl = (const __bf16*)smem;

  int lane = threadIdx.x & 31, wid = threadIdx.x >> 5;
  int nl = lane & 15, half = lane >> 4;
  int rb = blockIdx.x * 128 + wid * 16;
  int row = min(rb + nl, nrows - 1);
  const float* xr = X + (size_t)row * K;

  v8f acc[4];
#pragma unroll
  for (int c = 0; c < 4; ++c) acc[c] = zero8();
  for (int k0 = 0; k0 < K; k0 += 32) {
    v16bf a = loadA_f32(xr, k0, half);
#pragma unroll
    for (int c = 0; c < 4; ++c)
      acc[c] = wmma_bf(a, loadB_bf(Wl + (size_t)(c * 16 + nl) * K + k0 + half * 16), acc[c]);
  }
#pragma unroll
  for (int c = 0; c < 4; ++c) {
#pragma unroll
    for (int j = 0; j < 8; ++j) {
      int r = rb + j + half * 8;                 // C layout: M = j + 8*(lane>>4)
      if (r < nrows) Hm[(size_t)r * 64 + c * 16 + nl] = acc[c][j];
    }
  }
}

// ---------------------------------------------------------------------------
// attention logits per (node, head)
// ---------------------------------------------------------------------------
__global__ void k_logits(const float* h, const float* asrc, const float* adst,
                         float* als, float* ald, int n) {
  int t = blockIdx.x * blockDim.x + threadIdx.x;
  if (t >= n * 8) return;
  int node = t >> 3, hd = t & 7;
  const float* hp = h + (size_t)node * 64 + hd * 8;
  float s = 0.f, d = 0.f;
#pragma unroll
  for (int f = 0; f < 8; ++f) {
    float v = hp[f];
    s += v * asrc[hd * 8 + f];
    d += v * adst[hd * 8 + f];
  }
  als[t] = s;
  ald[t] = d;
}

__global__ void k_init(float* agg, int* mint, float* ssum, int n) {
  int t = blockIdx.x * blockDim.x + threadIdx.x;
  if (t < n * 64) agg[t] = 0.f;
  if (t < n * 8) { mint[t] = (int)0x80000000; ssum[t] = 0.f; }
}

// edge pass A: segment max of leaky-relu logits (native i32 atomic max)
__global__ void k_edge_max(const int* ei, int E, int n, const float* als,
                           const float* ald, int* mint) {
  int t = blockIdx.x * blockDim.x + threadIdx.x;
  if (t >= E + n) return;
  int s = t < E ? ei[t] : (t - E);
  int d = t < E ? ei[E + t] : (t - E);
#pragma unroll
  for (int hd = 0; hd < 8; ++hd) {
    float e = lrelu(als[s * 8 + hd] + ald[d * 8 + hd]);
    atomicMax(&mint[d * 8 + hd], fmap(e));
  }
}

// edge pass B: segment sum of exp(e - max)
__global__ void k_edge_sum(const int* ei, int E, int n, const float* als,
                           const float* ald, const int* mint, float* ssum) {
  int t = blockIdx.x * blockDim.x + threadIdx.x;
  if (t >= E + n) return;
  int s = t < E ? ei[t] : (t - E);
  int d = t < E ? ei[E + t] : (t - E);
#pragma unroll
  for (int hd = 0; hd < 8; ++hd) {
    float e = lrelu(als[s * 8 + hd] + ald[d * 8 + hd]);
    float ex = expf(e - funmap(mint[d * 8 + hd]));
    atomicAdd(&ssum[d * 8 + hd], ex);
  }
}

// edge pass C: agg[dst] += h[src] * alpha
__global__ void k_edge_agg(const int* ei, int E, int n, const float* als,
                           const float* ald, const int* mint, const float* ssum,
                           const float* h, float* agg) {
  int t = blockIdx.x * blockDim.x + threadIdx.x;
  if (t >= E + n) return;
  int s = t < E ? ei[t] : (t - E);
  int d = t < E ? ei[E + t] : (t - E);
  float al[8];
#pragma unroll
  for (int hd = 0; hd < 8; ++hd) {
    float e = lrelu(als[s * 8 + hd] + ald[d * 8 + hd]);
    al[hd] = expf(e - funmap(mint[d * 8 + hd])) / ssum[d * 8 + hd];
  }
  const float* hs = h + (size_t)s * 64;
  float* ag = agg + (size_t)d * 64;
#pragma unroll
  for (int c = 0; c < 64; ++c) atomicAdd(&ag[c], hs[c] * al[c >> 3]);
}

__global__ void k_finalize(const float* agg, const float* b, float* xo, int n, int do_elu) {
  int t = blockIdx.x * blockDim.x + threadIdx.x;
  if (t >= n * 64) return;
  float v = agg[t] + b[t & 63];
  if (do_elu) v = v > 0.f ? v : (expf(v) - 1.f);
  xo[t] = v;
}

// ---------------------------------------------------------------------------
// LSTM step: acc[16] tiles of [16 nodes x 256 gates] via WMMA; gate math and
// h/att reduction in registers. C/D tile layout: node M = j + 8*(lane>>4),
// gate col N = 16*t + (lane&15).
// ---------------------------------------------------------------------------
__device__ __forceinline__ void lstm_step(
    const float* xrow, const float* hprevrow,
    const __bf16* Wih, const __bf16* Whh,
    const float* bih, const float* bhh,
    v8f cst[4], float* hout, float* attdst,
    int l, bool accum, const float* watt, int waOff, int lane) {
  int half = lane >> 4, nl = lane & 15;
  v8f acc[16];
#pragma unroll
  for (int t = 0; t < 16; ++t) acc[t] = zero8();
#pragma unroll
  for (int k0 = 0; k0 < 64; k0 += 32) {
    v16bf aX = loadA_f32(xrow, k0, half);
    v16bf aH;
    if (hprevrow) aH = loadA_f32(hprevrow, k0, half);
#pragma unroll
    for (int t = 0; t < 16; ++t) {
      int col = t * 16 + nl;
      acc[t] = wmma_bf(aX, loadB_bf(Wih + (size_t)col * 64 + k0 + half * 16), acc[t]);
      if (hprevrow)
        acc[t] = wmma_bf(aH, loadB_bf(Whh + (size_t)col * 64 + k0 + half * 16), acc[t]);
    }
  }
  float attp[8];
#pragma unroll
  for (int j = 0; j < 8; ++j) attp[j] = 0.f;
#pragma unroll
  for (int t = 0; t < 4; ++t) {
    int ci = t * 16 + nl;                       // hidden channel, torch order i,f,g,o
    float bi = bih[ci] + bhh[ci];
    float bff = bih[64 + ci] + bhh[64 + ci];
    float bg = bih[128 + ci] + bhh[128 + ci];
    float bo = bih[192 + ci] + bhh[192 + ci];
    float wv = watt[waOff + ci];
#pragma unroll
    for (int j = 0; j < 8; ++j) {
      float iv = sigm(acc[t][j] + bi);
      float fv = sigm(acc[t + 4][j] + bff);
      float gv = tanhf(acc[t + 8][j] + bg);
      float ov = sigm(acc[t + 12][j] + bo);
      float cn = fv * cst[t][j] + iv * gv;
      cst[t][j] = cn;
      float hv = ov * tanhf(cn);
      hout[(j + half * 8) * 64 + ci] = hv;
      attp[j] += hv * wv;
    }
  }
#pragma unroll
  for (int j = 0; j < 8; ++j) {                 // reduce over the 16 N-lanes
    float v = attp[j];
#pragma unroll
    for (int m = 1; m < 16; m <<= 1) v += __shfl_xor(v, m, 16);
    if (nl == 0) {
      int mrow = j + half * 8;
      if (accum) attdst[mrow * 2 + l] += v;
      else       attdst[mrow * 2 + l]  = v;
    }
  }
}

// ---------------------------------------------------------------------------
// tail: bi-LSTM (L=2) + JK attention + output GEMM + log_softmax
// 128 threads = 4 waves, 16 nodes per wave.
// ---------------------------------------------------------------------------
__global__ void k_tail(const float* x1, const float* x2,
                       const __bf16* WihF, const __bf16* WhhF,
                       const __bf16* WihB, const __bf16* WhhB,
                       const float* bihF, const float* bhhF,
                       const float* bihB, const float* bhhB,
                       const float* watt, const __bf16* WoutP,
                       float* out, int n) {
  __shared__ float hA[4][1024];
  __shared__ float hB[4][1024];
  __shared__ float abuf[4][32];
  int lane = threadIdx.x & 31, wid = threadIdx.x >> 5;
  int nl = lane & 15, half = lane >> 4;
  int nb = blockIdx.x * 64 + wid * 16;
  int rowL = min(nb + nl, n - 1);
  const float* x1r = x1 + (size_t)rowL * 64;
  const float* x2r = x2 + (size_t)rowL * 64;

  v8f cst[4];
#pragma unroll
  for (int t = 0; t < 4; ++t) cst[t] = zero8();
  // forward: x1 -> l=0, x2 -> l=1
  lstm_step(x1r, nullptr, WihF, WhhF, bihF, bhhF, cst, hA[wid], abuf[wid], 0, false, watt, 0, lane);
  __syncthreads();
  lstm_step(x2r, &hA[wid][nl * 64], WihF, WhhF, bihF, bhhF, cst, hB[wid], abuf[wid], 1, false, watt, 0, lane);
  __syncthreads();
#pragma unroll
  for (int t = 0; t < 4; ++t) cst[t] = zero8();
  // backward: x2 -> l=1, then x1 -> l=0
  lstm_step(x2r, nullptr, WihB, WhhB, bihB, bhhB, cst, hA[wid], abuf[wid], 1, true, watt, 64, lane);
  __syncthreads();
  lstm_step(x1r, &hA[wid][nl * 64], WihB, WhhB, bihB, bhhB, cst, hB[wid], abuf[wid], 0, true, watt, 64, lane);
  __syncthreads();

  // alpha = softmax over L=2; final = alpha0*x1 + alpha1*x2, staged into hA
  float* fb = hA[wid];
  for (int idx = lane; idx < 1024; idx += 32) {
    int m = idx >> 6, ch = idx & 63;
    float a0 = abuf[wid][m * 2 + 0];
    float a1 = abuf[wid][m * 2 + 1];
    float mx = fmaxf(a0, a1);
    float e0 = expf(a0 - mx), e1 = expf(a1 - mx);
    float inv = 1.f / (e0 + e1);
    int r = min(nb + m, n - 1);
    fb[m * 64 + ch] = (e0 * x1[(size_t)r * 64 + ch] + e1 * x2[(size_t)r * 64 + ch]) * inv;
  }
  __syncthreads();

  // output GEMM: [16 x 64] @ [64 x 48(padded)] via WMMA
  v8f oacc[3];
#pragma unroll
  for (int t = 0; t < 3; ++t) oacc[t] = zero8();
  const float* frow = fb + nl * 64;
#pragma unroll
  for (int k0 = 0; k0 < 64; k0 += 32) {
    v16bf aF = loadA_f32(frow, k0, half);
#pragma unroll
    for (int t = 0; t < 3; ++t)
      oacc[t] = wmma_bf(aF, loadB_bf(WoutP + (size_t)(t * 16 + nl) * 64 + k0 + half * 16), oacc[t]);
  }
  // log_softmax over 40 valid classes (cols 40..47 masked)
  float rmax[8], rls[8];
#pragma unroll
  for (int j = 0; j < 8; ++j) {
    float v0 = oacc[0][j], v1 = oacc[1][j];
    float v2 = (nl < 8) ? oacc[2][j] : -3.4e38f;
    float mv = fmaxf(fmaxf(v0, v1), v2);
#pragma unroll
    for (int m = 1; m < 16; m <<= 1) mv = fmaxf(mv, __shfl_xor(mv, m, 16));
    float s = expf(v0 - mv) + expf(v1 - mv) + ((nl < 8) ? expf(v2 - mv) : 0.f);
#pragma unroll
    for (int m = 1; m < 16; m <<= 1) s += __shfl_xor(s, m, 16);
    rmax[j] = mv;
    rls[j] = logf(s);
  }
#pragma unroll
  for (int t = 0; t < 3; ++t) {
    int col = t * 16 + nl;
#pragma unroll
    for (int j = 0; j < 8; ++j) {
      int r = nb + j + half * 8;
      if (col < 40 && r < n)
        out[(size_t)r * 40 + col] = oacc[t][j] - rmax[j] - rls[j];
    }
  }
}

// ---------------------------------------------------------------------------
// launch
// ---------------------------------------------------------------------------
extern "C" void kernel_launch(void* const* d_in, const int* in_sizes, int n_in,
                              void* d_out, int out_size, void* d_ws, size_t ws_size,
                              hipStream_t stream) {
  const float* x    = (const float*)d_in[0];
  const int*   ei   = (const int*)d_in[1];
  const float* W1   = (const float*)d_in[2];
  const float* a1s  = (const float*)d_in[3];
  const float* a1d  = (const float*)d_in[4];
  const float* b1   = (const float*)d_in[5];
  const float* W2   = (const float*)d_in[6];
  const float* a2s  = (const float*)d_in[7];
  const float* a2d  = (const float*)d_in[8];
  const float* b2   = (const float*)d_in[9];
  const float* WihF = (const float*)d_in[10];
  const float* WhhF = (const float*)d_in[11];
  const float* bihF = (const float*)d_in[12];
  const float* bhhF = (const float*)d_in[13];
  const float* WihB = (const float*)d_in[14];
  const float* WhhB = (const float*)d_in[15];
  const float* bihB = (const float*)d_in[16];
  const float* bhhB = (const float*)d_in[17];
  const float* watt = (const float*)d_in[18];
  const float* Wout = (const float*)d_in[20];
  float* out = (float*)d_out;
  (void)n_in; (void)out_size; (void)ws_size;

  int N = in_sizes[0] / 512;
  int E = in_sizes[1] / 2;
  int ET = E + N;

  char* base = (char*)d_ws;
  size_t off = 0;
  auto alloc = [&](size_t bytes) -> void* {
    off = (off + 255) & ~(size_t)255;
    void* p = base + off;
    off += bytes;
    return p;
  };
  __bf16* w1t   = (__bf16*)alloc(64 * 512 * 2);
  __bf16* w2t   = (__bf16*)alloc(64 * 64 * 2);
  __bf16* wihFb = (__bf16*)alloc(256 * 64 * 2);
  __bf16* whhFb = (__bf16*)alloc(256 * 64 * 2);
  __bf16* wihBb = (__bf16*)alloc(256 * 64 * 2);
  __bf16* whhBb = (__bf16*)alloc(256 * 64 * 2);
  __bf16* woutp = (__bf16*)alloc(48 * 64 * 2);
  float* hbuf = (float*)alloc((size_t)N * 64 * 4);
  float* als  = (float*)alloc((size_t)N * 8 * 4);
  float* ald  = (float*)alloc((size_t)N * 8 * 4);
  int*   mint = (int*)alloc((size_t)N * 8 * 4);
  float* ssum = (float*)alloc((size_t)N * 8 * 4);
  float* agg  = (float*)alloc((size_t)N * 64 * 4);
  float* x1   = (float*)alloc((size_t)N * 64 * 4);
  float* x2   = (float*)alloc((size_t)N * 64 * 4);

  dim3 blk(256);
  int preptot = 64 * 512 + 64 * 64 + 4 * 16384 + 48 * 64;
  k_prep<<<(preptot + 255) / 256, blk, 0, stream>>>(W1, W2, WihF, WhhF, WihB, WhhB, Wout,
                                                    w1t, w2t, wihFb, whhFb, wihBb, whhBb, woutp);

  int gblocks = (N + 127) / 128;
  // ---- GAT layer 1 ----
  k_gemm<<<gblocks, 256, 64 * 512 * 2, stream>>>(x, w1t, hbuf, N, 512);
  k_logits<<<((N * 8) + 255) / 256, blk, 0, stream>>>(hbuf, a1s, a1d, als, ald, N);
  k_init<<<((N * 64) + 255) / 256, blk, 0, stream>>>(agg, mint, ssum, N);
  k_edge_max<<<(ET + 255) / 256, blk, 0, stream>>>(ei, E, N, als, ald, mint);
  k_edge_sum<<<(ET + 255) / 256, blk, 0, stream>>>(ei, E, N, als, ald, mint, ssum);
  k_edge_agg<<<(ET + 255) / 256, blk, 0, stream>>>(ei, E, N, als, ald, mint, ssum, hbuf, agg);
  k_finalize<<<((N * 64) + 255) / 256, blk, 0, stream>>>(agg, b1, x1, N, 1);

  // ---- GAT layer 2 ----
  k_gemm<<<gblocks, 256, 64 * 64 * 2, stream>>>(x1, w2t, hbuf, N, 64);
  k_logits<<<((N * 8) + 255) / 256, blk, 0, stream>>>(hbuf, a2s, a2d, als, ald, N);
  k_init<<<((N * 64) + 255) / 256, blk, 0, stream>>>(agg, mint, ssum, N);
  k_edge_max<<<(ET + 255) / 256, blk, 0, stream>>>(ei, E, N, als, ald, mint);
  k_edge_sum<<<(ET + 255) / 256, blk, 0, stream>>>(ei, E, N, als, ald, mint, ssum);
  k_edge_agg<<<(ET + 255) / 256, blk, 0, stream>>>(ei, E, N, als, ald, mint, ssum, hbuf, agg);
  k_finalize<<<((N * 64) + 255) / 256, blk, 0, stream>>>(agg, b2, x2, N, 0);

  // ---- JK bi-LSTM + attention + classifier ----
  k_tail<<<(N + 63) / 64, 128, 0, stream>>>(x1, x2, wihFb, whhFb, wihBb, whhBb,
                                            bihF, bhhF, bihB, bhhB, watt, woutp, out, N);
}
